// BlockBasedLocalAttention_24343874634045
// MI455X (gfx1250) — matrix operand order repaired and checked by reading, not compiled, not run on hardware
//
#include <hip/hip_runtime.h>
#include <hip/hip_bf16.h>

// ---------------------------------------------------------------------------
// Block-based local attention for MI455X (gfx1250), bf16 WMMA pipeline.
//   K0: wconv     : {Wq,Wk,Wv,Wo}(f32) -> transposed bf16 copies (once)
//   K1: qkv_proj  : x(f32) @ W -> q,k (bf16 row-major), v^T (bf16)
//                   weights staged to LDS via TENSOR_LOAD_TO_LDS (TDM)
//   K2: local_attn: K context staged via TDM (contiguous spans), V^T context
//                   staged via TDM 2D (unclipped) or async-to-LDS (edges),
//                   logits=QK^T/16 in LDS, softmax, out = A @ V (WMMA)
//   K3: out_proj  : attn_out(bf16) @ Wo + bo -> f32 output
// Intermediates (~168MB bf16) fit in the 192MB L2.
// ---------------------------------------------------------------------------

typedef __attribute__((ext_vector_type(16))) __bf16 v16bf;
typedef __attribute__((ext_vector_type(8)))  __bf16 v8bf;
typedef __attribute__((ext_vector_type(8)))  float  v8f;
typedef __attribute__((ext_vector_type(4)))  unsigned int u32x4;
typedef __attribute__((ext_vector_type(8)))  int i32x8;
typedef __attribute__((ext_vector_type(4)))  int i32x4;

#define EDIM  256
#define SDIM  4096
#define BLK   128
#define NB    (SDIM / BLK)   // 32 blocks per sequence
#define BATCH 20
#define CTX   (3 * BLK)      // 384 context positions per block
#define CTXH  (CTX / 2)      // 192: context half staged in LDS

union FragCvt { v16bf v; v8bf h[2]; };

// A-matrix 16x32 bf16 fragment (CDNA5 layout):
//  lanes 0-15: row m0+lane, K = k0+{0..7} and k0+{16..23}
//  lanes 16-31: row m0+lane-16, K = k0+{8..15} and k0+{24..31}
__device__ inline v16bf frag_a_load(const __bf16* base, int m0, int k0, int ld, int lane) {
  const int m  = m0 + (lane & 15);
  const int kb = k0 + ((lane < 16) ? 0 : 8);
  const __bf16* p = base + (long)m * ld + kb;
  FragCvt f;
  f.h[0] = *(const v8bf*)(p);
  f.h[1] = *(const v8bf*)(p + 16);
  return f.v;
}

// B-matrix 32x16 bf16 fragment; baseT is K-major (column n contiguous in K):
//  lanes 0-15: col n0+lane, K = k0+{0..7} and k0+{16..23} ; lanes 16-31: col n0+lane-16, K = k0+{8..15} and k0+{24..31}
__device__ inline v16bf frag_b_load(const __bf16* baseT, int n0, long k0, long ld, int lane) {
  const int  n  = n0 + (lane & 15);
  const long kb = k0 + ((lane < 16) ? 0 : 8);
  const __bf16* p = baseT + (long)n * ld + kb;
  FragCvt f;
  f.h[0] = *(const v8bf*)(p);
  f.h[1] = *(const v8bf*)(p + 16);
  return f.v;
}

__device__ inline v8f wmma_bf16(v16bf a, v16bf b, v8f c) {
  return __builtin_amdgcn_wmma_f32_16x16x32_bf16(false, a, false, b, (short)0, c,
                                                 false, false);
}

// Async global->LDS copy (ASYNCcnt path). loff is the absolute LDS byte
// offset (dynamic-LDS base is 0: no static __shared__ in these kernels).
__device__ inline void async_copy_b128(unsigned loff, const void* gptr) {
  asm volatile("global_load_async_to_lds_b128 %0, %1, off"
               :: "v"(loff), "v"(gptr) : "memory");
}
__device__ inline void wait_async0() {
  asm volatile("s_wait_asynccnt 0x0" ::: "memory");
}

// TDM 2D copy global -> LDS (TENSORcnt path).  Moves `rows` rows of `w8`
// 8-byte elements, source row stride `stride8` (8B units); rows pack
// contiguously in LDS (stride w8*8 bytes).  D# packing per ISA 08 §8.
// Use rows==1, stride8==w8 for a flat 1D span.
__device__ inline void tdm_copy_2d(unsigned lds_off, const void* gsrc,
                                   unsigned w8, unsigned rows, unsigned stride8) {
  const unsigned long long ga = (unsigned long long)gsrc;
  u32x4 g0;
  g0[0] = 1u;                                          // count=1 (valid user D#)
  g0[1] = lds_off;                                     // lds_addr (bytes)
  g0[2] = (unsigned)ga;                                // global_addr[31:0]
  g0[3] = (unsigned)((ga >> 32) & 0x01FFFFFFu)         // global_addr[56:32]
          | (2u << 30);                                // type=2 ("image")
  i32x8 g1;
  g1[0] = (int)(3u << 16);                             // data_size=8B, no flags
  g1[1] = (int)((w8 & 0xFFFFu) << 16);                 // tensor_dim0[15:0]
  g1[2] = (int)(((w8 >> 16) & 0xFFFFu)                 // tensor_dim0[31:16]
          | ((rows & 0xFFFFu) << 16));                 // tensor_dim1[15:0]
  g1[3] = (int)(((rows >> 16) & 0xFFFFu)               // tensor_dim1[31:16]
          | ((w8 & 0xFFFFu) << 16));                   // tile_dim0 (<=65535)
  g1[4] = (int)(rows & 0xFFFFu);                       // tile_dim1, tile_dim2=0
  g1[5] = (int)stride8;                                // tensor_dim0_stride lo
  g1[6] = 0;                                           // stride0 hi, stride1 lo
  g1[7] = 0;                                           // stride1 hi
  i32x4 z4 = {};
#if defined(__clang_major__) && (__clang_major__ >= 23)
  i32x8 z8 = {};
  __builtin_amdgcn_tensor_load_to_lds(g0, g1, z4, z4, z8, 0);
#else
  __builtin_amdgcn_tensor_load_to_lds(g0, g1, z4, z4, 0);
#endif
}
__device__ inline void wait_tensor0() {
  __builtin_amdgcn_s_wait_tensorcnt((short)0);
}

// ---------------------------------------------------------------------------
// Kernel 0: one-time weight convert + transpose to bf16 (K-major columns).
// ---------------------------------------------------------------------------
__global__ __launch_bounds__(256) void wconv_kernel(
    const float* __restrict__ Wq, const float* __restrict__ Wk,
    const float* __restrict__ Wv, const float* __restrict__ Wo,
    __bf16* __restrict__ qT, __bf16* __restrict__ kT,
    __bf16* __restrict__ vT, __bf16* __restrict__ oT) {
  const float* src[4] = {Wq, Wk, Wv, Wo};
  __bf16*      dst[4] = {qT, kT, vT, oT};
  const int i = blockIdx.x * 256 + threadIdx.x;   // 65536 threads == E*E
  const int n = i >> 8, d = i & 255;
#pragma unroll
  for (int m = 0; m < 4; ++m)
    dst[m][i] = (__bf16)src[m][d * EDIM + n];     // wT[n][d] = W[d][n]
}

// ---------------------------------------------------------------------------
// Kernel 1: QKV projection.  grid = 640 row-blocks of 128, block = 256 (8 waves)
// LDS: x stripe bf16 (64KB) + W^T bf16 (128KB, TDM-staged) = 192KB
// ---------------------------------------------------------------------------
__global__ __launch_bounds__(256) void qkv_proj_kernel(
    const float* __restrict__ x,
    const __bf16* __restrict__ wqT, const __bf16* __restrict__ wkT,
    const __bf16* __restrict__ wvT,
    __bf16* __restrict__ qo, __bf16* __restrict__ ko, __bf16* __restrict__ vTo) {
  extern __shared__ char smem[];
  __bf16* xs = (__bf16*)smem;                                  // 128 x 256
  __bf16* wT = (__bf16*)(smem + BLK * EDIM * 2);               // 256 x 256
  const unsigned wtoff = BLK * EDIM * 2;                       // LDS offset of wT
  const int tid = threadIdx.x, lane = tid & 31, wave = tid >> 5;
  const long row0 = (long)blockIdx.x * BLK;

  for (int i = tid; i < BLK * EDIM; i += 256)
    xs[i] = (__bf16)x[row0 * EDIM + i];

  const __bf16* wsrc[3] = {wqT, wkT, wvT};
  for (int w = 0; w < 3; ++w) {
    __syncthreads();                 // previous GEMM done before re-staging wT
    if (wave == 0) {                 // one TDM descriptor moves all 128KB
      tdm_copy_2d(wtoff, wsrc[w], EDIM * EDIM / 4, 1u, EDIM * EDIM / 4);
      wait_tensor0();
    }
    __syncthreads();
    const int m0 = wave * 16;
    for (int nt = 0; nt < EDIM / 16; ++nt) {
      v8f acc = {};
#pragma unroll
      for (int kk = 0; kk < EDIM; kk += 32) {
        const v16bf a  = frag_a_load(xs, m0, kk, EDIM, lane);
        const v16bf bm = frag_b_load(wT, nt * 16, kk, EDIM, lane);
        acc = wmma_bf16(a, bm, acc);
      }
      const int col = nt * 16 + (lane & 15);
      const int mh  = m0 + ((lane < 16) ? 0 : 8);
      if (w < 2) {
        __bf16* dst = (w == 0) ? qo : ko;
#pragma unroll
        for (int j = 0; j < 8; ++j)
          dst[(row0 + mh + j) * EDIM + col] = (__bf16)acc[j];
      } else {
        // v stored transposed: vT[b][e][s] -> AV GEMM B-frags contiguous in s
#pragma unroll
        for (int j = 0; j < 8; ++j) {
          const long m = row0 + mh + j;
          const long b = m >> 12, s = m & (SDIM - 1);
          vTo[(b * EDIM + col) * SDIM + s] = (__bf16)acc[j];
        }
      }
    }
  }
}

// ---------------------------------------------------------------------------
// Kernel 2: local attention.  grid = BATCH*NB = 640, block = 256 (8 waves)
// LDS: logits 128x384 f32 (192KB) + K/V context half stage (96KB) = 288KB
// ---------------------------------------------------------------------------
#define LOGITS_BYTES (BLK * CTX * 4)     // 196608

__global__ __launch_bounds__(256) void local_attn_kernel(
    const __bf16* __restrict__ q, const __bf16* __restrict__ k,
    const __bf16* __restrict__ vT, __bf16* __restrict__ ao) {
  extern __shared__ char smem[];
  float*  logits  = (float*)smem;                          // row stride CTX f32
  __bf16* kvstage = (__bf16*)(smem + LOGITS_BYTES);        // 96KB shared stage
  const unsigned kvoff = LOGITS_BYTES;
  const int tid = threadIdx.x, lane = tid & 31, wave = tid >> 5;
  const int nb = blockIdx.x & (NB - 1);
  const int b  = blockIdx.x / NB;
  const long qrow0 = (long)b * SDIM + (long)nb * BLK;
  const int ctx0 = nb * BLK - BLK;                         // may be negative
  const int m0 = wave * 16;
  const __bf16* qbase = q  + qrow0 * EDIM;
  const __bf16* kbase = k  + (long)b * SDIM * EDIM;
  const __bf16* vbase = vT + (long)b * EDIM * SDIM;

  // ---- Phase 1: logits = (Q K^T)/16, context processed in two halves ----
  for (int h = 0; h < 2; ++h) {
    if (h) __syncthreads();            // previous half consumed before restage
    // K rows are contiguous in memory -> clipped span is one TDM 1D copy
    const int base = ctx0 + h * CTXH;
    const int r0v = (base < 0) ? -base : 0;
    const int r1v = (base + CTXH > SDIM) ? (SDIM - base) : CTXH;
    if (wave == 0 && r1v > r0v) {
      const unsigned n8 = (unsigned)(r1v - r0v) * (EDIM / 4);  // rows*512B/8
      tdm_copy_2d(kvoff + (unsigned)r0v * (EDIM * 2),
                  kbase + (long)(base + r0v) * EDIM, n8, 1u, n8);
      wait_tensor0();
    }
    __syncthreads();
    for (int ntl = 0; ntl < CTXH / 16; ++ntl) {            // 12 tiles per half
      const int nt  = h * (CTXH / 16) + ntl;
      const int p0  = ctx0 + nt * 16;
      const int col = nt * 16 + (lane & 15);
      const int mh  = m0 + ((lane < 16) ? 0 : 8);
      if (p0 >= 0 && p0 < SDIM) {      // uniform branch, EXEC stays full
        v8f acc = {};
#pragma unroll
        for (int kk = 0; kk < EDIM; kk += 32) {
          const v16bf a  = frag_a_load(qbase, m0, kk, EDIM, lane);
          const v16bf bm = frag_b_load(kvstage, ntl * 16, kk, EDIM, lane);
          acc = wmma_bf16(a, bm, acc);
        }
#pragma unroll
        for (int j = 0; j < 8; ++j)
          logits[(mh + j) * CTX + col] = acc[j] * 0.0625f;
      } else {                          // zero-padded K -> logit == 0
#pragma unroll
        for (int j = 0; j < 8; ++j)
          logits[(mh + j) * CTX + col] = 0.0f;
      }
    }
  }
  __syncthreads();

  // ---- Phase 2: row softmax, rewrite row in-place as bf16 ----
  if (tid < BLK) {
    float* rowp = logits + tid * CTX;
    float mx = rowp[0];
    for (int i = 1; i < CTX; ++i) mx = fmaxf(mx, rowp[i]);
    float sum = 0.f;
    for (int i = 0; i < CTX; ++i) { const float e = __expf(rowp[i] - mx); rowp[i] = e; sum += e; }
    const float inv = 1.f / sum;
    __bf16* arow = (__bf16*)rowp;       // ascending in-place rewrite is safe
    for (int i = 0; i < CTX; ++i) arow[i] = (__bf16)(rowp[i] * inv);
  }

  // ---- Phase 3: out = A @ V, V^T context staged per half, acc held live ----
  const __bf16* attn = (const __bf16*)logits;   // row stride 2*CTX bf16 elems
  v8f acc[EDIM / 16];
#pragma unroll
  for (int et = 0; et < EDIM / 16; ++et) acc[et] = (v8f){};

  for (int h = 0; h < 2; ++h) {
    __syncthreads();                   // kvstage free / attn ready
    const int base = ctx0 + h * CTXH;
    const bool clipped = (base < 0) || (base + CTXH > SDIM);
    if (!clipped) {
      // clean 2D tile: 256 rows x 384B, src stride 8KB -> one TDM descriptor
      if (wave == 0) {
        tdm_copy_2d(kvoff, vbase + base, CTXH / 4, EDIM, SDIM / 4);
        wait_tensor0();
      }
    } else {
      // edge blocks: per-lane clipped async copies (ASYNCcnt path)
      for (int c = tid; c < EDIM * (CTXH / 8); c += 256) { // 6144 x 16B chunks
        const int e = c / 24, ch = c - e * 24;
        const int p = base + ch * 8;
        if (p >= 0 && p + 7 < SDIM)
          async_copy_b128(kvoff + (unsigned)(e * CTXH + ch * 8) * 2,
                          vbase + (long)e * SDIM + p);
      }
      wait_async0();
    }
    __syncthreads();
#pragma unroll
    for (int et = 0; et < EDIM / 16; ++et) {
#pragma unroll
      for (int kk = 0; kk < CTXH; kk += 32) {
        const int p = base + kk;
        if (p >= 0 && p < SDIM) {      // uniform; padded V==0 -> skip
          const v16bf a  = frag_a_load(attn, m0, h * CTXH + kk, 2 * CTX, lane);
          const v16bf bm = frag_b_load(kvstage, et * 16, kk, CTXH, lane);
          acc[et] = wmma_bf16(a, bm, acc[et]);
        }
      }
    }
  }

  __bf16* orow = ao + qrow0 * EDIM;
#pragma unroll
  for (int et = 0; et < EDIM / 16; ++et) {
    const int col = et * 16 + (lane & 15);
    const int mh  = m0 + ((lane < 16) ? 0 : 8);
#pragma unroll
    for (int j = 0; j < 8; ++j)
      orow[(mh + j) * EDIM + col] = (__bf16)acc[et][j];
  }
}

// ---------------------------------------------------------------------------
// Kernel 3: output projection + bias.  grid = 640, block = 256
// LDS: Wo^T bf16 (128KB, TDM-staged)
// ---------------------------------------------------------------------------
__global__ __launch_bounds__(256) void out_proj_kernel(
    const __bf16* __restrict__ ain, const __bf16* __restrict__ woT,
    const float* __restrict__ bo, float* __restrict__ out) {
  extern __shared__ char smem[];
  __bf16* wT = (__bf16*)smem;
  const int tid = threadIdx.x, lane = tid & 31, wave = tid >> 5;
  const long row0 = (long)blockIdx.x * BLK;
  if (wave == 0) {
    tdm_copy_2d(0u, woT, EDIM * EDIM / 4, 1u, EDIM * EDIM / 4);
    wait_tensor0();
  }
  __syncthreads();
  const __bf16* abase = ain + row0 * EDIM;
  const int m0 = wave * 16;
  for (int nt = 0; nt < EDIM / 16; ++nt) {
    v8f acc = {};
#pragma unroll
    for (int kk = 0; kk < EDIM; kk += 32) {
      const v16bf a  = frag_a_load(abase, m0, kk, EDIM, lane);
      const v16bf bm = frag_b_load(wT, nt * 16, kk, EDIM, lane);
      acc = wmma_bf16(a, bm, acc);
    }
    const int col = nt * 16 + (lane & 15);
    const int mh  = m0 + ((lane < 16) ? 0 : 8);
    const float bias = bo[col];
#pragma unroll
    for (int j = 0; j < 8; ++j)
      out[(row0 + mh + j) * EDIM + col] = acc[j] + bias;
  }
  (void)tid;
}

// ---------------------------------------------------------------------------
extern "C" void kernel_launch(void* const* d_in, const int* in_sizes, int n_in,
                              void* d_out, int out_size, void* d_ws, size_t ws_size,
                              hipStream_t stream) {
  (void)in_sizes; (void)n_in; (void)out_size; (void)ws_size;
  const float* x  = (const float*)d_in[0];
  const float* Wq = (const float*)d_in[1];
  const float* Wk = (const float*)d_in[2];
  const float* Wv = (const float*)d_in[3];
  const float* Wo = (const float*)d_in[4];
  const float* bo = (const float*)d_in[5];
  float* out = (float*)d_out;

  char* ws = (char*)d_ws;
  const size_t tokBytes = (size_t)BATCH * SDIM * EDIM * sizeof(__bf16); // ~41.9MB
  const size_t wBytes   = (size_t)EDIM * EDIM * sizeof(__bf16);         // 128KB
  __bf16* qb  = (__bf16*)(ws);
  __bf16* kb  = (__bf16*)(ws + tokBytes);
  __bf16* vTb = (__bf16*)(ws + 2 * tokBytes);
  __bf16* ab  = (__bf16*)(ws + 3 * tokBytes);
  __bf16* wqT = (__bf16*)(ws + 4 * tokBytes);
  __bf16* wkT = (__bf16*)(ws + 4 * tokBytes + wBytes);
  __bf16* wvT = (__bf16*)(ws + 4 * tokBytes + 2 * wBytes);
  __bf16* woT = (__bf16*)(ws + 4 * tokBytes + 3 * wBytes);

  const int nRowBlocks = BATCH * SDIM / BLK;          // 640
  const int lds1 = BLK * EDIM * 2 + EDIM * EDIM * 2;  // 192KB
  const int lds2 = LOGITS_BYTES + CTXH * EDIM * 2;    // 288KB
  const int lds3 = EDIM * EDIM * 2;                   // 128KB
  hipFuncSetAttribute((const void*)qkv_proj_kernel,
                      hipFuncAttributeMaxDynamicSharedMemorySize, lds1);
  hipFuncSetAttribute((const void*)local_attn_kernel,
                      hipFuncAttributeMaxDynamicSharedMemorySize, lds2);
  hipFuncSetAttribute((const void*)out_proj_kernel,
                      hipFuncAttributeMaxDynamicSharedMemorySize, lds3);

  wconv_kernel<<<EDIM * EDIM / 256, 256, 0, stream>>>(Wq, Wk, Wv, Wo,
                                                      wqT, wkT, wvT, woT);
  qkv_proj_kernel<<<nRowBlocks, 256, lds1, stream>>>(x, wqT, wkT, wvT, qb, kb, vTb);
  local_attn_kernel<<<BATCH * NB, 256, lds2, stream>>>(qb, kb, vTb, ab);
  out_proj_kernel<<<nRowBlocks, 256, lds3, stream>>>(ab, woT, bo, out);
}
